// TinyYOLOv3_27058293964839
// MI455X (gfx1250) — matrix operand order repaired and checked by reading, not compiled, not run on hardware
//
#include <hip/hip_runtime.h>
#include <hip/hip_bf16.h>

typedef __attribute__((ext_vector_type(16))) _Float16 v16h;
typedef __attribute__((ext_vector_type(8)))  float    v8f;

#define ALPHA   0.1f
#define EPS_BN  1e-3f
#define NPRED   2535
#define MAX_OUT 10
#define IOU_T   0.6f
#define SCORE_T 0.5f

// ---------------------------------------------------------------- elementwise
__global__ void cast_f32_to_f16_kernel(const float* __restrict__ in,
                                       _Float16* __restrict__ out, int n) {
  int i = blockIdx.x * blockDim.x + threadIdx.x;
  if (i < n) out[i] = (_Float16)in[i];
}

// Pack HWIO f32 weights into the WMMA B-fragment lane layout:
// per 32-K chunk: 512 f16 = 32 lanes x 16 halves; lane<16 holds K=0..7,16..23 (N=lane),
// lane>=16 holds K=8..15,24..31 (N=lane-16).  (CDNA5 ISA §7.12.2)
__global__ void prep_w_kernel(const float* __restrict__ w, _Float16* __restrict__ packed,
                              int Ktot, int Cout, int kChunks, int nTiles) {
  int idx = blockIdx.x * blockDim.x + threadIdx.x;
  int total = nTiles * kChunks * 512;
  if (idx >= total) return;
  int nT   = idx / (kChunks * 512);
  int r    = idx % (kChunks * 512);
  int ch   = r / 512;
  int q    = r % 512;
  int lane = q / 16;
  int s    = q % 16;
  int kLocal = (s < 8) ? ((lane >> 4) * 8 + s) : (16 + (lane >> 4) * 8 + (s - 8));
  int n = nT * 16 + (lane & 15);
  int k = ch * 32 + kLocal;
  float v = (k < Ktot && n < Cout) ? w[k * Cout + n] : 0.0f;
  packed[idx] = (_Float16)v;
}

// ---------------------------------------------------------------- WMMA conv
union H8 { uint4 u; _Float16 h[8]; };

// Implicit-GEMM conv: M = B*H*W, N = Cout, K = KHW*KHW*CIN.
// Each wave computes a 16M x 32N tile: one A fragment feeds two WMMAs.
// CIN/KHW are compile-time so all K->(kh,kw,cin) math strength-reduces.
template <int CIN, int KHW>
__global__ __launch_bounds__(256)
void conv_wmma_kernel(const _Float16* __restrict__ in, const _Float16* __restrict__ wpack,
                      const float* __restrict__ bias, float* __restrict__ out,
                      int Bn, int H, int W, int Cout) {
  constexpr int KW = KHW, P = KHW >> 1;
  constexpr int Ktot = KHW * KHW * CIN;
  constexpr int kChunks = (Ktot + 31) / 32;
  constexpr bool fastA = (CIN % 8) == 0;

  const int M = Bn * H * W;
  const int lane = threadIdx.x & 31;
  const int wave = threadIdx.x >> 5;
  const int mTile = (blockIdx.x * 8 + wave) * 16;
  if (mTile >= M) return;          // uniform per wave
  const int nTile0 = blockIdx.y * 2;

  // A-fragment: lane (l&15) owns output row mTile+(l&15); lane<16 needs K runs
  // [0..7],[16..23] of each chunk, lane>=16 needs [8..15],[24..31].
  const int mrow = mTile + (lane & 15);
  const bool mv = mrow < M;
  int bb = 0, oh = 0, ow = 0;
  if (mv) { bb = mrow / (H * W); int pix = mrow % (H * W); oh = pix / W; ow = pix % W; }

  v8f acc0 = {}, acc1 = {};
  const _Float16* b0 = wpack + ((size_t)nTile0 * kChunks) * 512 + (size_t)lane * 16;
  const _Float16* b1 = b0 + (size_t)kChunks * 512;

#pragma unroll 4
  for (int ch = 0; ch < kChunks; ++ch) {
    const int k0 = ch * 32;
    v16h a;
#pragma unroll
    for (int g = 0; g < 2; ++g) {
      const int kb = k0 + ((lane & 16) ? 8 : 0) + g * 16;
      if (fastA) {
        H8 v; v.u = make_uint4(0u, 0u, 0u, 0u);
        if (mv && kb < Ktot) {
          int slice = kb / CIN, cin = kb % CIN;     // compile-time strength-reduced
          int kh = slice / KW, kw = slice % KW;
          int ih = oh + kh - P, iw = ow + kw - P;
          if (ih >= 0 && ih < H && iw >= 0 && iw < W)
            v.u = *(const uint4*)(in + (((size_t)bb * H + ih) * W + iw) * CIN + cin);
        }
#pragma unroll
        for (int s2 = 0; s2 < 8; ++s2) a[g * 8 + s2] = v.h[s2];
      } else {                      // only conv1 (CIN=3, Ktot=27)
#pragma unroll
        for (int s2 = 0; s2 < 8; ++s2) {
          int k = kb + s2;
          _Float16 hv = (_Float16)0.f;
          if (mv && k < Ktot) {
            int slice = k / CIN, cin = k % CIN;
            int kh = slice / KW, kw = slice % KW;
            int ih = oh + kh - P, iw = ow + kw - P;
            if (ih >= 0 && ih < H && iw >= 0 && iw < W)
              hv = in[(((size_t)bb * H + ih) * W + iw) * CIN + cin];
          }
          a[g * 8 + s2] = hv;
        }
      }
    }
    v16h bf0 = *(const v16h*)(b0 + (size_t)ch * 512);
    v16h bf1 = *(const v16h*)(b1 + (size_t)ch * 512);
    if (ch + 1 < kChunks) __builtin_prefetch(b0 + (size_t)(ch + 1) * 512, 0, 1);
    acc0 = __builtin_amdgcn_wmma_f32_16x16x32_f16(false, a, false, bf0,
                                                  (short)0, acc0, false, false);
    acc1 = __builtin_amdgcn_wmma_f32_16x16x32_f16(false, a, false, bf1,
                                                  (short)0, acc1, false, false);
  }

  // D layout: VGPR r holds M=r (lanes 0..15) / M=8+r (lanes 16..31), N=lane&15.
  const int mb = mTile + ((lane & 16) ? 8 : 0);
  {
    const int n = nTile0 * 16 + (lane & 15);
    if (n < Cout) {
      const float bsv = bias ? bias[n] : 0.0f;
#pragma unroll
      for (int r = 0; r < 8; ++r) {
        int m = mb + r;
        if (m < M) out[(size_t)m * Cout + n] = acc0[r] + bsv;
      }
    }
  }
  {
    const int n = (nTile0 + 1) * 16 + (lane & 15);
    if (n < Cout) {
      const float bsv = bias ? bias[n] : 0.0f;
#pragma unroll
      for (int r = 0; r < 8; ++r) {
        int m = mb + r;
        if (m < M) out[(size_t)m * Cout + n] = acc1[r] + bsv;
      }
    }
  }
}

// ---------------------------------------------------------------- batchnorm
__global__ void stats_kernel(const float* __restrict__ x, float* __restrict__ stats,
                             int N, int C) {
  __shared__ float ssum[256], ssq[256];
  const int c = blockIdx.x;
  float s = 0.f, q = 0.f;
  for (int i = threadIdx.x; i < N; i += 256) {
    float v = x[(size_t)i * C + c];
    s += v; q += v * v;
  }
  ssum[threadIdx.x] = s; ssq[threadIdx.x] = q;
  __syncthreads();
  for (int off = 128; off > 0; off >>= 1) {
    if ((int)threadIdx.x < off) {
      ssum[threadIdx.x] += ssum[threadIdx.x + off];
      ssq[threadIdx.x]  += ssq[threadIdx.x + off];
    }
    __syncthreads();
  }
  if (threadIdx.x == 0) {
    float mean = ssum[0] / (float)N;
    float var  = ssq[0] / (float)N - mean * mean;
    stats[c] = mean;
    stats[1024 + c] = var > 0.f ? var : 0.f;
  }
}

__global__ void bn_leaky_kernel(const float* __restrict__ x, const float* __restrict__ stats,
                                const float* __restrict__ gamma, const float* __restrict__ beta,
                                _Float16* __restrict__ out, int N, int C) {
  int i = blockIdx.x * blockDim.x + threadIdx.x;
  if (i >= N * C) return;
  int c = i % C;
  float sc = gamma[c] * rsqrtf(stats[1024 + c] + EPS_BN);
  float y = (x[i] - stats[c]) * sc + beta[c];
  y = y > 0.f ? y : ALPHA * y;
  out[i] = (_Float16)y;
}

// ---------------------------------------------------------------- pool / upcat
__global__ void pool_kernel(const _Float16* __restrict__ in, _Float16* __restrict__ out,
                            int Bn, int Hin, int Win, int C, int stride, int Hout, int Wout) {
  int i = blockIdx.x * blockDim.x + threadIdx.x;
  if (i >= Bn * Hout * Wout * C) return;
  int c = i % C; int r = i / C;
  int ow = r % Wout; r /= Wout;
  int oh = r % Hout; int b = r / Hout;
  float best = -INFINITY;
#pragma unroll
  for (int dy = 0; dy < 2; ++dy)
#pragma unroll
    for (int dx = 0; dx < 2; ++dx) {
      int ih = oh * stride + dy, iw = ow * stride + dx;
      float v = (ih < Hin && iw < Win)
                  ? (float)in[(((size_t)b * Hin + ih) * Win + iw) * C + c] : 0.0f;
      best = best > v ? best : v;
    }
  out[i] = (_Float16)best;
}

__global__ void upcat_kernel(const _Float16* __restrict__ u, const _Float16* __restrict__ root,
                             _Float16* __restrict__ out, int Bn) {
  int i = blockIdx.x * blockDim.x + threadIdx.x;
  if (i >= Bn * 26 * 26 * 384) return;
  int c = i % 384; int r = i / 384;
  int ow = r % 26; r /= 26; int oh = r % 26; int b = r / 26;
  _Float16 v;
  if (c < 128) v = u[(((size_t)b * 13 + (oh >> 1)) * 13 + (ow >> 1)) * 128 + c];
  else         v = root[(((size_t)b * 26 + oh) * 26 + ow) * 256 + (c - 128)];
  out[i] = v;
}

// ---------------------------------------------------------------- decode / NMS
__global__ void decode_kernel(const float* __restrict__ head, float* __restrict__ boxes,
                              float* __restrict__ scores, int Bn, int gs, int rowOff,
                              float aw0, float ah0, float aw1, float ah1, float aw2, float ah2) {
  int rows = gs * gs * 3;
  int i = blockIdx.x * blockDim.x + threadIdx.x;
  if (i >= Bn * rows) return;
  int b = i / rows, p = i % rows;
  int pix = p / 3, a = p % 3;
  int oh = pix / gs, ow = pix % gs;
  const float* t = head + (((size_t)b * gs + oh) * gs + ow) * 15 + a * 5;
  float aw = a == 0 ? aw0 : (a == 1 ? aw1 : aw2);
  float ah = a == 0 ? ah0 : (a == 1 ? ah1 : ah2);
  float cx = (1.f / (1.f + __expf(-t[0])) + (float)ow) / (float)gs;
  float cy = (1.f / (1.f + __expf(-t[1])) + (float)oh) / (float)gs;
  float wv = __expf(t[2]) * aw;
  float hv = __expf(t[3]) * ah;
  float ob = 1.f / (1.f + __expf(-t[4]));
  int row = rowOff + p;
  float* bo = boxes + ((size_t)b * NPRED + row) * 4;
  bo[0] = cx - wv * 0.5f; bo[1] = cy - hv * 0.5f;
  bo[2] = cx + wv * 0.5f; bo[3] = cy + hv * 0.5f;
  scores[(size_t)b * NPRED + row] = ob;
}

__global__ __launch_bounds__(256)
void nms_kernel(const float* __restrict__ boxes, const float* __restrict__ scores0,
                float* __restrict__ outp) {
  __shared__ float s[NPRED];
  __shared__ float area[NPRED];
  __shared__ float redV[256];
  __shared__ int   redI[256];
  __shared__ int   selIdx[MAX_OUT];
  __shared__ float selSc[MAX_OUT];
  __shared__ int   selV[MAX_OUT];
  __shared__ float cb[4];
  __shared__ float curArea;
  __shared__ int   curI;
  __shared__ int   curValid;
  const int tid = threadIdx.x;
  for (int j = tid; j < NPRED; j += 256) {
    float sc = scores0[j];
    s[j] = (sc >= SCORE_T) ? sc : -INFINITY;
    const float* bj = boxes + (size_t)j * 4;
    area[j] = (bj[2] - bj[0]) * (bj[3] - bj[1]);
  }
  __syncthreads();
  for (int t = 0; t < MAX_OUT; ++t) {
    float bv = -INFINITY; int bi = NPRED;
    for (int j = tid; j < NPRED; j += 256) {
      float v = s[j];
      if (v > bv || (v == bv && j < bi)) { bv = v; bi = j; }
    }
    redV[tid] = bv; redI[tid] = bi;
    __syncthreads();
    for (int off = 128; off > 0; off >>= 1) {
      if (tid < off) {
        float v2 = redV[tid + off]; int i2 = redI[tid + off];
        if (v2 > redV[tid] || (v2 == redV[tid] && i2 < redI[tid])) {
          redV[tid] = v2; redI[tid] = i2;
        }
      }
      __syncthreads();
    }
    if (tid == 0) {
      int i = redI[0]; if (i >= NPRED) i = 0;   // jnp.argmax of all -inf -> 0
      float si = s[i];
      int valid = (si > -INFINITY) ? 1 : 0;
      curI = i; curValid = valid;
      selIdx[t] = i; selSc[t] = valid ? si : 0.f; selV[t] = valid;
      const float* bx = boxes + (size_t)i * 4;
      cb[0] = bx[0]; cb[1] = bx[1]; cb[2] = bx[2]; cb[3] = bx[3];
      curArea = area[i];
    }
    __syncthreads();
    if (curValid) {
      int i = curI;
      for (int j = tid; j < NPRED; j += 256) {
        const float* bj = boxes + (size_t)j * 4;
        float iw = fminf(bj[2], cb[2]) - fmaxf(bj[0], cb[0]); iw = iw > 0.f ? iw : 0.f;
        float ih = fminf(bj[3], cb[3]) - fmaxf(bj[1], cb[1]); ih = ih > 0.f ? ih : 0.f;
        float inter = iw * ih;
        float iou = inter / (area[j] + curArea - inter + 1e-9f);
        if (iou > IOU_T || j == i) s[j] = -INFINITY;
      }
    }
    __syncthreads();
  }
  for (int e = tid; e < 8 * MAX_OUT * 4; e += 256) {
    int b = e / (MAX_OUT * 4);
    int rem = e % (MAX_OUT * 4);
    int tt = rem / 4, c = rem % 4;
    outp[e] = boxes[((size_t)b * NPRED + selIdx[tt]) * 4 + c];
  }
  if (tid < MAX_OUT) {
    outp[320 + tid] = selSc[tid];
    outp[330 + tid] = selV[tid] ? 1.0f : 0.0f;
  }
}

// ---------------------------------------------------------------- host
template <int CIN, int KHW>
static void launch_conv(const _Float16* inA, const float* w, const float* bias, float* outC,
                        int Bn, int H, int Wd, int Cout, _Float16* WPK, hipStream_t stream) {
  constexpr int Ktot = KHW * KHW * CIN;
  constexpr int kChunks = (Ktot + 31) / 32;
  const int nPairs = (Cout + 31) / 32;
  const int nTiles = nPairs * 2;                 // pad to even so 2nd frag is in-bounds
  const int totalPack = nTiles * kChunks * 512;
  prep_w_kernel<<<(totalPack + 255) / 256, 256, 0, stream>>>(w, WPK, Ktot, Cout,
                                                             kChunks, nTiles);
  const int M = Bn * H * Wd;
  dim3 grid((M + 127) / 128, nPairs);
  conv_wmma_kernel<CIN, KHW><<<grid, 256, 0, stream>>>(inA, WPK, bias, outC,
                                                       Bn, H, Wd, Cout);
}

extern "C" void kernel_launch(void* const* d_in, const int* in_sizes, int n_in,
                              void* d_out, int out_size, void* d_ws, size_t ws_size,
                              hipStream_t stream) {
  (void)in_sizes; (void)n_in; (void)out_size; (void)ws_size;
  const float* x = (const float*)d_in[0];
  auto W_  = [&](int l) { return (const float*)d_in[1 + 3 * l]; };
  auto G_  = [&](int l) { return (const float*)d_in[2 + 3 * l]; };
  auto Bt_ = [&](int l) { return (const float*)d_in[3 + 3 * l]; };
  const float* h1w = (const float*)d_in[34];
  const float* h1b = (const float*)d_in[35];
  const float* h2w = (const float*)d_in[36];
  const float* h2b = (const float*)d_in[37];

  char* ws = (char*)d_ws;
  size_t off = 0;
  auto alloc = [&](size_t bytes) -> char* {
    char* p = ws + off;
    off += (bytes + 255) & ~(size_t)255;
    return p;
  };
  const int Bn = 8;
  _Float16* AIN   = (_Float16*)alloc((size_t)Bn * 416 * 416 * 3 * 2);
  float*    CONV  = (float*)   alloc((size_t)Bn * 416 * 416 * 16 * 4);
  _Float16* ACT0  = (_Float16*)alloc((size_t)Bn * 416 * 416 * 16 * 2);
  _Float16* ACT1  = (_Float16*)alloc((size_t)Bn * 416 * 416 * 16 * 2);
  _Float16* ROOT  = (_Float16*)alloc((size_t)Bn * 26 * 26 * 256 * 2);
  _Float16* ROUTE = (_Float16*)alloc((size_t)Bn * 13 * 13 * 256 * 2);
  _Float16* CAT   = (_Float16*)alloc((size_t)Bn * 26 * 26 * 384 * 2);
  _Float16* WPK   = (_Float16*)alloc((size_t)10 * 1024 * 1024);  // max c7 pack 9.44MB
  float*    STATS = (float*)   alloc(2048 * 4);
  float*    HEAD1 = (float*)   alloc((size_t)Bn * 169 * 15 * 4);
  float*    HEAD2 = (float*)   alloc((size_t)Bn * 676 * 15 * 4);
  float*    BOXES = (float*)   alloc((size_t)Bn * NPRED * 4 * 4);
  float*    SCORES= (float*)   alloc((size_t)Bn * NPRED * 4);

  auto bnact = [&](const float* c, int layer, _Float16* outA, int N, int C) {
    stats_kernel<<<C, 256, 0, stream>>>(c, STATS, N, C);
    int total = N * C;
    bn_leaky_kernel<<<(total + 255) / 256, 256, 0, stream>>>(c, STATS, G_(layer), Bt_(layer),
                                                             outA, N, C);
  };
  auto pool = [&](const _Float16* inA, _Float16* outA, int Hin, int Win, int C,
                  int stride, int Hout, int Wout) {
    int total = Bn * Hout * Wout * C;
    pool_kernel<<<(total + 255) / 256, 256, 0, stream>>>(inA, outA, Bn, Hin, Win, C,
                                                         stride, Hout, Wout);
  };

  { int n = Bn * 416 * 416 * 3;
    cast_f32_to_f16_kernel<<<(n + 255) / 256, 256, 0, stream>>>(x, AIN, n); }

  launch_conv<3, 3>(AIN, W_(0), nullptr, CONV, Bn, 416, 416, 16, WPK, stream);
  bnact(CONV, 0, ACT0, Bn * 416 * 416, 16);
  pool(ACT0, ACT1, 416, 416, 16, 2, 208, 208);

  launch_conv<16, 3>(ACT1, W_(1), nullptr, CONV, Bn, 208, 208, 32, WPK, stream);
  bnact(CONV, 1, ACT0, Bn * 208 * 208, 32);
  pool(ACT0, ACT1, 208, 208, 32, 2, 104, 104);

  launch_conv<32, 3>(ACT1, W_(2), nullptr, CONV, Bn, 104, 104, 64, WPK, stream);
  bnact(CONV, 2, ACT0, Bn * 104 * 104, 64);
  pool(ACT0, ACT1, 104, 104, 64, 2, 52, 52);

  launch_conv<64, 3>(ACT1, W_(3), nullptr, CONV, Bn, 52, 52, 128, WPK, stream);
  bnact(CONV, 3, ACT0, Bn * 52 * 52, 128);
  pool(ACT0, ACT1, 52, 52, 128, 2, 26, 26);

  launch_conv<128, 3>(ACT1, W_(4), nullptr, CONV, Bn, 26, 26, 256, WPK, stream);
  bnact(CONV, 4, ROOT, Bn * 26 * 26, 256);           // keep: route to upsample concat
  pool(ROOT, ACT1, 26, 26, 256, 2, 13, 13);

  launch_conv<256, 3>(ACT1, W_(5), nullptr, CONV, Bn, 13, 13, 512, WPK, stream);
  bnact(CONV, 5, ACT0, Bn * 13 * 13, 512);
  pool(ACT0, ACT1, 13, 13, 512, 1, 13, 13);          // pad + stride-1 maxpool

  launch_conv<512, 3>(ACT1, W_(6), nullptr, CONV, Bn, 13, 13, 1024, WPK, stream);
  bnact(CONV, 6, ACT0, Bn * 13 * 13, 1024);

  launch_conv<1024, 1>(ACT0, W_(7), nullptr, CONV, Bn, 13, 13, 256, WPK, stream);
  bnact(CONV, 7, ROUTE, Bn * 13 * 13, 256);          // keep: route

  launch_conv<256, 3>(ROUTE, W_(8), nullptr, CONV, Bn, 13, 13, 512, WPK, stream);
  bnact(CONV, 8, ACT0, Bn * 13 * 13, 512);

  launch_conv<512, 1>(ACT0, h1w, h1b, HEAD1, Bn, 13, 13, 15, WPK, stream); // head1

  launch_conv<256, 1>(ROUTE, W_(9), nullptr, CONV, Bn, 13, 13, 128, WPK, stream);
  bnact(CONV, 9, ACT1, Bn * 13 * 13, 128);

  { int total = Bn * 26 * 26 * 384;
    upcat_kernel<<<(total + 255) / 256, 256, 0, stream>>>(ACT1, ROOT, CAT, Bn); }

  launch_conv<384, 3>(CAT, W_(10), nullptr, CONV, Bn, 26, 26, 256, WPK, stream);
  bnact(CONV, 10, ACT0, Bn * 26 * 26, 256);

  launch_conv<256, 1>(ACT0, h2w, h2b, HEAD2, Bn, 26, 26, 15, WPK, stream);  // head2

  { int total = Bn * 13 * 13 * 3;
    decode_kernel<<<(total + 255) / 256, 256, 0, stream>>>(HEAD1, BOXES, SCORES, Bn, 13, 0,
        0.22f, 0.44f, 0.35f, 0.68f, 0.55f, 0.90f); }
  { int total = Bn * 26 * 26 * 3;
    decode_kernel<<<(total + 255) / 256, 256, 0, stream>>>(HEAD2, BOXES, SCORES, Bn, 26, 507,
        0.05f, 0.12f, 0.10f, 0.22f, 0.16f, 0.35f); }

  nms_kernel<<<1, 256, 0, stream>>>(BOXES, SCORES, (float*)d_out);
}
